// GMM_56014963475035
// MI455X (gfx1250) — matrix operand order repaired and checked by reading, not compiled, not run on hardware
//
#include <hip/hip_runtime.h>
#include <hip/hip_bf16.h>
#include <math.h>

// ---------------- problem constants ----------------
#define N_CLUSTERS 16
#define DIM 128           // latent dim (GEMM K)
#define MM 100            // interpolation steps
#define BATCH 8192
#define N_STATES 136      // 16*17/2
#define NCOL (N_STATES * MM)        // 13600 columns of the virtual [B, S*M] matrix
#define NTILE (NCOL / 16)           // 850 column tiles of 16
// -0.5*D*log(2*pi) - log(M)
#define KCONST (-122.2293024f)

// LDS staging geometry: one tile = 16 columns, each column = 256B hi + 256B lo
// contiguous in global (512B), stored in LDS with a 16B pad per 512B (TDM
// pad_interval=512B, pad_amount=16B) -> column stride 528B = 132 DW == 4 mod 64
// (conflict-minimal ds_load_b128).
#define COL_G_BYTES 512
#define COL_L_BYTES 528
#define TILE_G_BYTES (16 * COL_G_BYTES)   // 8192
#define TILE_L_BYTES (16 * COL_L_BYTES)   // 8448

typedef __attribute__((ext_vector_type(16))) __bf16 bf16x16;
typedef __attribute__((ext_vector_type(8)))  float  f32x8;
typedef __attribute__((ext_vector_type(4)))  unsigned int u32x4;
typedef __attribute__((ext_vector_type(4)))  int i32x4;
typedef __attribute__((ext_vector_type(8)))  int i32x8;

union BFrag {
    bf16x16 v;
    uint4   q[2];
};

// ---------------------------------------------------------------
// TDM: 1-D tile copy, 1024 x 8B elements, LDS pad 16B per 512B.
// D# packing per ISA §8.3/8.4; 6-arg builtin (clang-23 / therock form).
// ---------------------------------------------------------------
__device__ __forceinline__ void tdm_tile_load(const void* gsrc, unsigned lds_off) {
    unsigned long long ga = (unsigned long long)(size_t)gsrc;
    u32x4 g0 = { 1u,                                   // count=1 (valid user D#)
                 lds_off,                              // lds_addr (bytes)
                 (unsigned)(ga & 0xFFFFFFFFu),         // global_addr[31:0]
                 ((unsigned)((ga >> 32) & 0x01FFFFFFull)) | 0x80000000u }; // [56:32] | type=2
    // word0: data_size=3 (8B) | pad_enable | pad_interval=6 (512B) | pad_amount=3 (16B)
    i32x8 g1 = { 0x07930000,
                 (int)(1024u << 16),   // tensor_dim0[15:0]=1024 in bits 63:48
                 0x00010000,           // tensor_dim1=1 in bits 95:80
                 (int)(1024u << 16),   // tile_dim0=1024 in bits 127:112
                 1,                    // tile_dim1=1
                 1024,                 // tensor_dim0_stride
                 0, 0 };
    i32x4 gz4 = { 0, 0, 0, 0 };
    i32x8 gz8 = { 0, 0, 0, 0, 0, 0, 0, 0 };
    __builtin_amdgcn_tensor_load_to_lds(g0, g1, gz4, gz4, gz8, 0);
}

// ---------------------------------------------------------------
// head: softmax(pi) -> out[0:136], log_pi -> ws, copy mu -> out[136:2184]
// ---------------------------------------------------------------
__global__ __launch_bounds__(256) void head_kernel(const float* __restrict__ pi,
                                                   const float* __restrict__ mu,
                                                   float* __restrict__ out,
                                                   float* __restrict__ logpi) {
    int tid = threadIdx.x;
    if (tid < N_STATES) {
        float mx = -3.4e38f;
        for (int i = 0; i < N_STATES; ++i) mx = fmaxf(mx, pi[i]);
        float s = 0.f;
        for (int i = 0; i < N_STATES; ++i) s += __expf(pi[i] - mx);
        float p = __expf(pi[tid] - mx) / s;
        out[tid]   = p;
        logpi[tid] = __logf(p + 1e-30f);
    }
    for (int i = tid; i < DIM * N_CLUSTERS; i += 256) out[N_STATES + i] = mu[i];
}

// ---------------------------------------------------------------
// zprep: one block per batch row.  z -> bf16 hi/lo split + ||z||^2
// ---------------------------------------------------------------
__global__ __launch_bounds__(128) void zprep_kernel(const float* __restrict__ z,
                                                    __bf16* __restrict__ zh,
                                                    __bf16* __restrict__ zl,
                                                    float* __restrict__ z2) {
    __shared__ float part[4];
    int b = blockIdx.x, d = threadIdx.x;
    float v  = z[(size_t)b * DIM + d];
    __bf16 h = (__bf16)v;
    __bf16 l = (__bf16)(v - (float)h);
    zh[(size_t)b * DIM + d] = h;
    zl[(size_t)b * DIM + d] = l;
    float sq = v * v;
    for (int off = 16; off > 0; off >>= 1) sq += __shfl_down(sq, off, 32);
    if ((threadIdx.x & 31) == 0) part[threadIdx.x >> 5] = sq;
    __syncthreads();
    if (threadIdx.x == 0) z2[b] = part[0] + part[1] + part[2] + part[3];
}

// ---------------------------------------------------------------
// mcol: one block per column c = s*100 + m.  Interpolated mean, stored
// column-major as [c][hi 128 bf16][lo 128 bf16] (512B per column, so one
// 16-column tile is one contiguous 8KB TDM copy), plus ||m||^2 and log_pi.
// ---------------------------------------------------------------
__global__ __launch_bounds__(128) void mcol_kernel(const float* __restrict__ mu,
                                                   const float* __restrict__ w,
                                                   const int* __restrict__ A,
                                                   const int* __restrict__ B,
                                                   const float* __restrict__ logpi,
                                                   __bf16* __restrict__ mhl,
                                                   float* __restrict__ m2,
                                                   float* __restrict__ lpc) {
    __shared__ float part[4];
    int c  = blockIdx.x;
    int s  = c / MM;
    int mi = c - s * MM;
    int d  = threadIdx.x;
    float wv  = w[mi];
    int   ia  = A[s], ib = B[s];
    float val = mu[d * N_CLUSTERS + ia] * wv + mu[d * N_CLUSTERS + ib] * (1.0f - wv);
    __bf16 h = (__bf16)val;
    __bf16 l = (__bf16)(val - (float)h);
    mhl[(size_t)c * 256 + d]       = h;
    mhl[(size_t)c * 256 + 128 + d] = l;
    float sq = val * val;
    for (int off = 16; off > 0; off >>= 1) sq += __shfl_down(sq, off, 32);
    if ((threadIdx.x & 31) == 0) part[threadIdx.x >> 5] = sq;
    __syncthreads();
    if (threadIdx.x == 0) {
        m2[c]  = part[0] + part[1] + part[2] + part[3];
        lpc[c] = logpi[s];
    }
}

// ---------------------------------------------------------------
// main fused GEMM + online logsumexp.
// 64 blocks x 8 waves; each wave owns 16 batch rows, block shares each
// B tile via TDM -> LDS double buffer (8x cut in L2 traffic vs per-wave
// streaming). 12 x v_wmma_f32_16x16x32_bf16 per tile (bf16x3 f32 emulation).
// ---------------------------------------------------------------
__global__ __launch_bounds__(256) void gmm_main(const float* __restrict__ z2,
                                                const __bf16* __restrict__ zh,
                                                const __bf16* __restrict__ zl,
                                                const __bf16* __restrict__ mhl,
                                                const float* __restrict__ m2,
                                                const float* __restrict__ lpc,
                                                float* __restrict__ outlp) {
    __shared__ __align__(16) unsigned char smem[2 * TILE_L_BYTES];

    const int lane    = threadIdx.x & 31;
    const int wave    = threadIdx.x >> 5;
    const int rowBase = blockIdx.x * 128 + wave * 16;
    const bool upper  = lane >= 16;
    const int  l16    = lane & 15;

    // ---- resident A fragments (ISA 7.12.2 16-bit A layout) ----
    // lanes 0-15 : u[0..3] = K 0..7,  u[4..7] = K 16..23  (per 32-K block)
    // lanes 16-31: u[0..3] = K 8..15, u[4..7] = K 24..31
    BFrag ah[4], al[4];
    {
        const __bf16* zrh = zh + (size_t)(rowBase + l16) * DIM;
        const __bf16* zrl = zl + (size_t)(rowBase + l16) * DIM;
#pragma unroll
        for (int kb = 0; kb < 4; ++kb) {
            int dA = kb * 32 + (upper ? 8 : 0);
            int dB = kb * 32 + (upper ? 24 : 16);
            ah[kb].q[0] = *(const uint4*)(zrh + dA);
            ah[kb].q[1] = *(const uint4*)(zrh + dB);
            al[kb].q[0] = *(const uint4*)(zrl + dA);
            al[kb].q[1] = *(const uint4*)(zrl + dB);
        }
    }

    // C/D layout: VGPR v -> row M = v (lanes 0-15) or 8+v (lanes 16-31); N = l16
    float z2v[8], mx[8], sm[8];
#pragma unroll
    for (int v = 0; v < 8; ++v) {
        z2v[v] = z2[rowBase + v + (upper ? 8 : 0)];
        mx[v]  = -INFINITY;
        sm[v]  = 0.0f;
    }

    // prologue: stage tile 0 into buffer 0
    if (wave == 0) {
        tdm_tile_load(mhl, (unsigned)(size_t)&smem[0]);
        __builtin_amdgcn_s_wait_tensorcnt(0);
    }
    __syncthreads();

    for (int t = 0; t < NTILE; ++t) {
        // kick TDM for the next tile into the alternate buffer (its previous
        // readers all passed the t-1 barrier)
        if (wave == 0 && t + 1 < NTILE) {
            tdm_tile_load((const unsigned char*)mhl + (size_t)(t + 1) * TILE_G_BYTES,
                          (unsigned)(size_t)&smem[((t + 1) & 1) * TILE_L_BYTES]);
        }

        const int c = t * 16 + l16;
        const unsigned char* colh = &smem[(t & 1) * TILE_L_BYTES] + l16 * COL_L_BYTES;
        const unsigned char* coll = colh + 256;

        f32x8 acc = {0.f, 0.f, 0.f, 0.f, 0.f, 0.f, 0.f, 0.f};
#pragma unroll
        for (int kb = 0; kb < 4; ++kb) {
            // B layout: lanes 0-15 hold K 0..7,16..23, lanes 16-31 hold K 8..15,24..31
            const int d0 = kb * 32 + (upper ? 8 : 0);
            BFrag bh, bl;
            bh.q[0] = *(const uint4*)(colh + d0 * 2);
            bh.q[1] = *(const uint4*)(colh + d0 * 2 + 32);
            bl.q[0] = *(const uint4*)(coll + d0 * 2);
            bl.q[1] = *(const uint4*)(coll + d0 * 2 + 32);
            acc = __builtin_amdgcn_wmma_f32_16x16x32_bf16(false, ah[kb].v, false, bh.v,
                                                          (short)0, acc, false, false);
            acc = __builtin_amdgcn_wmma_f32_16x16x32_bf16(false, ah[kb].v, false, bl.v,
                                                          (short)0, acc, false, false);
            acc = __builtin_amdgcn_wmma_f32_16x16x32_bf16(false, al[kb].v, false, bh.v,
                                                          (short)0, acc, false, false);
        }

        const float m2c = m2[c];
        const float lp  = lpc[c];
#pragma unroll
        for (int v = 0; v < 8; ++v) {
            float sq  = z2v[v] - 2.0f * acc[v] + m2c;
            float val = lp - 0.5f * sq;
            float nm  = fmaxf(mx[v], val);
            sm[v] = sm[v] * __expf(mx[v] - nm) + __expf(val - nm);
            mx[v] = nm;
        }

        // publish the prefetched buffer: TDM done on wave 0, then barrier
        if (wave == 0) __builtin_amdgcn_s_wait_tensorcnt(0);
        __syncthreads();
    }

    // merge (max,sum) across the 16 lanes of each half-wave (columns)
#pragma unroll
    for (int off = 1; off < 16; off <<= 1) {
#pragma unroll
        for (int v = 0; v < 8; ++v) {
            float om = __shfl_xor(mx[v], off, 32);
            float os = __shfl_xor(sm[v], off, 32);
            float nm = fmaxf(mx[v], om);
            sm[v] = sm[v] * __expf(mx[v] - nm) + os * __expf(om - nm);
            mx[v] = nm;
        }
    }

    if (l16 == 0) {
#pragma unroll
        for (int v = 0; v < 8; ++v)
            outlp[rowBase + v + (upper ? 8 : 0)] = mx[v] + __logf(sm[v]) + KCONST;
    }
}

// ---------------------------------------------------------------
extern "C" void kernel_launch(void* const* d_in, const int* in_sizes, int n_in,
                              void* d_out, int out_size, void* d_ws, size_t ws_size,
                              hipStream_t stream) {
    const float* z  = (const float*)d_in[0];
    const float* mu = (const float*)d_in[1];
    const float* pi = (const float*)d_in[2];
    const float* w  = (const float*)d_in[3];
    const int*   A  = (const int*)d_in[4];
    const int*   B  = (const int*)d_in[5];
    float* out = (float*)d_out;

    // workspace carve (256B-aligned chunks), ~11.2 MB total
    char* p = (char*)d_ws;
    auto carve = [&](size_t bytes) { char* r = p; p += (bytes + 255) & ~(size_t)255; return r; };
    __bf16* z_hi = (__bf16*)carve((size_t)BATCH * DIM * sizeof(__bf16));
    __bf16* z_lo = (__bf16*)carve((size_t)BATCH * DIM * sizeof(__bf16));
    __bf16* mhl  = (__bf16*)carve((size_t)NCOL * 256 * sizeof(__bf16));
    float* z2    = (float*)carve((size_t)BATCH * sizeof(float));
    float* m2    = (float*)carve((size_t)NCOL * sizeof(float));
    float* lpcv  = (float*)carve((size_t)NCOL * sizeof(float));
    float* logpi = (float*)carve((size_t)N_STATES * sizeof(float));

    head_kernel<<<1, 256, 0, stream>>>(pi, mu, out, logpi);
    zprep_kernel<<<BATCH, 128, 0, stream>>>(z, z_hi, z_lo, z2);
    mcol_kernel<<<NCOL, 128, 0, stream>>>(mu, w, A, B, logpi, mhl, m2, lpcv);
    gmm_main<<<BATCH / 128, 256, 0, stream>>>(z2, z_hi, z_lo, mhl, m2, lpcv,
                                              out + N_STATES + DIM * N_CLUSTERS);
}
